// DCRNN_83073257439394
// MI455X (gfx1250) — matrix operand order repaired and checked by reading, not compiled, not run on hardware
//
#include <hip/hip_runtime.h>
#include <math.h>

// ---------------------------------------------------------------------------
// DCRNN forward for MI455X (gfx1250, wave32, WMMA).
// Core compute: bf16 WMMA (f32 accumulate) batched GEMMs, L2-resident supports.
// Each GEMM wave computes a 32x16 D tile: two 16x16 WMMAs sharing one
// B-fragment per K=32 step (3 x 32B loads : 2 wmma).
// ---------------------------------------------------------------------------

typedef __attribute__((ext_vector_type(16))) __bf16 v16bf;
typedef __attribute__((ext_vector_type(8)))  __bf16 v8bf;
typedef __attribute__((ext_vector_type(8)))  float  v8f;

#define NB    512        // nodes
#define BB    32         // batch
#define BN    (BB * NB)  // 16384
#define UNITS 64
#define MDIFF 5          // diffusion terms (2 supports * K=2 + identity)
#define HORIZON 12

static __device__ __forceinline__ v16bf cat8(v8bf lo, v8bf hi) {
  return __builtin_shufflevector(lo, hi, 0,1,2,3,4,5,6,7,8,9,10,11,12,13,14,15);
}

// ---------------------------------------------------------------------------
// Setup kernels
// ---------------------------------------------------------------------------

// Per (b,i): 1/rowsum(adj+I) and 1/colsum(adj+I)  (colsum = rowsum of adj^T)
__global__ __launch_bounds__(256) void dinv_kernel(
    const float* __restrict__ adj, float* __restrict__ dinv1,
    float* __restrict__ dinv2) {
  int idx = blockIdx.x * blockDim.x + threadIdx.x;
  if (idx >= BN) return;
  int b = idx >> 9, i = idx & (NB - 1);
  const float* base = adj + (size_t)b * NB * NB;
  float rs = 1.f, cs = 1.f;  // +I diagonal contribution
  for (int j = 0; j < NB; ++j) {
    rs += base[(size_t)i * NB + j];
    cs += base[(size_t)j * NB + i];
  }
  dinv1[idx] = 1.f / rs;
  dinv2[idx] = 1.f / cs;
}

// A1 = D1^-1 (adj+I),  A2 = D2^-1 (adj^T+I), both bf16 row-major
__global__ __launch_bounds__(256) void build_supports_kernel(
    const float* __restrict__ adj, const float* __restrict__ dinv1,
    const float* __restrict__ dinv2, __bf16* __restrict__ A1,
    __bf16* __restrict__ A2) {
  size_t idx = (size_t)blockIdx.x * blockDim.x + threadIdx.x;
  if (idx >= (size_t)BB * NB * NB) return;
  int j = (int)(idx & (NB - 1));
  int i = (int)((idx >> 9) & (NB - 1));
  int b = (int)(idx >> 18);
  float diag = (i == j) ? 1.f : 0.f;
  float a1 = (adj[((size_t)b * NB + i) * NB + j] + diag) * dinv1[b * NB + i];
  float a2 = (adj[((size_t)b * NB + j) * NB + i] + diag) * dinv2[b * NB + i];
  A1[idx] = (__bf16)a1;
  A2[idx] = (__bf16)a2;
}

// Wt[out][Kp] = W[k][out] (bf16, zero-padded to Kp)
__global__ __launch_bounds__(256) void build_wt_kernel(
    const float* __restrict__ W, __bf16* __restrict__ Wt, int R, int Cc,
    int Kp) {
  int idx = blockIdx.x * blockDim.x + threadIdx.x;
  if (idx >= Cc * Kp) return;
  int c = idx / Kp, k = idx % Kp;
  Wt[idx] = (__bf16)((k < R) ? W[(size_t)k * Cc + c] : 0.f);
}

__global__ __launch_bounds__(256) void zero_f32_kernel(float* __restrict__ p,
                                                       long n) {
  long idx = (long)blockIdx.x * blockDim.x + threadIdx.x;
  if (idx < n) p[idx] = 0.f;
}

// Zero the ragged-K pad columns of xcat (cols [fM, Kp) of every node row)
__global__ __launch_bounds__(256) void zero_xcat_pad_kernel(
    __bf16* __restrict__ xcat, int Kp, int fM, int padc) {
  int idx = blockIdx.x * blockDim.x + threadIdx.x;
  if (idx >= BN * padc) return;
  int n = idx / padc, j = idx % padc;
  xcat[(size_t)n * Kp + fM + j] = (__bf16)0.f;
}

// ---------------------------------------------------------------------------
// gconv input assembly: x0 = concat(x_in, state) -> fp32 node-major (padded),
// bf16 feature-major Xt0 (WMMA B-operand), and xcat column m=0.
// ---------------------------------------------------------------------------
__global__ __launch_bounds__(256) void prepare_x0_kernel(
    const float* __restrict__ xin, int xin_bstride, int cin,
    const float* __restrict__ state,  // [BN][64]
    float* __restrict__ x0nm,         // [BN][f_pad]
    __bf16* __restrict__ Xt0,         // [B][f_pad][NB]
    __bf16* __restrict__ xcat,        // [BN][Kp]
    int Kp, int f, int f_pad) {
  int idx = blockIdx.x * blockDim.x + threadIdx.x;
  if (idx >= BN * f_pad) return;
  int fp = idx % f_pad;
  int bn = idx / f_pad;
  int b = bn >> 9, n = bn & (NB - 1);
  float v = 0.f;
  if (fp < cin)
    v = xin[(size_t)b * xin_bstride + (size_t)n * cin + fp];
  else if (fp < f)
    v = state[(size_t)bn * UNITS + (fp - cin)];
  x0nm[(size_t)bn * f_pad + fp] = v;
  Xt0[((size_t)b * f_pad + fp) * NB + n] = (__bf16)v;
  if (fp < f) xcat[(size_t)bn * Kp + fp * MDIFF + 0] = (__bf16)v;
}

// ---------------------------------------------------------------------------
// Batched diffusion GEMM: D = alpha * A(512x512) * X(512xf) + beta * Y
//   A  : bf16 [B][512][512]   (A-fragment: two contiguous 16B loads/lane)
//   Xt : bf16 [B][f_pad][512] (B-fragment: two contiguous 16B loads/lane)
// One wave = 32x16 D tile (two stacked 16x16 WMMAs sharing the B-fragment).
// Epilogue writes: out_nm fp32 (coalesced), xcat bf16 (interleaved col),
// optional XtOut bf16 (next GEMM's operand).  Beta/XtOut are compile-time.
// ---------------------------------------------------------------------------
template <bool HAS_BETA, bool HAS_XTOUT>
__global__ __launch_bounds__(256) void gemm_diff_kernel(
    const __bf16* __restrict__ A, const __bf16* __restrict__ Xt,
    float* __restrict__ out_nm, const float* __restrict__ betaSrc, float alpha,
    float beta, __bf16* __restrict__ xcat, int Kp, int m, int f, int f_pad,
    __bf16* __restrict__ XtOut, int totalTiles) {
  int wid = (blockIdx.x * blockDim.x + threadIdx.x) >> 5;
  if (wid >= totalTiles) return;  // wave-uniform: EXEC stays all-ones
  int lane = threadIdx.x & 31;
  int hb = lane >> 4;  // lane half-group
  int lr = lane & 15;
  int nt_f = f_pad >> 4;
  int b = wid / (16 * nt_f);   // 16 row-tiles of 32
  int r = wid % (16 * nt_f);
  int mt = r / nt_f, nt = r % nt_f;

  const __bf16* pa0 = A + ((size_t)b * NB + mt * 32 + lr) * NB + hb * 8;
  const __bf16* pa1 = pa0 + (size_t)16 * NB;
  const __bf16* pb = Xt + ((size_t)b * f_pad + nt * 16 + lr) * NB + hb * 8;

  v8f acc0 = {0.f, 0.f, 0.f, 0.f, 0.f, 0.f, 0.f, 0.f};
  v8f acc1 = {0.f, 0.f, 0.f, 0.f, 0.f, 0.f, 0.f, 0.f};
#pragma unroll 4
  for (int k = 0; k < NB; k += 32) {
    v16bf bv = cat8(*(const v8bf*)(pb + k), *(const v8bf*)(pb + k + 16));
    v16bf a0 = cat8(*(const v8bf*)(pa0 + k), *(const v8bf*)(pa0 + k + 16));
    v16bf a1 = cat8(*(const v8bf*)(pa1 + k), *(const v8bf*)(pa1 + k + 16));
    __builtin_prefetch(pa0 + k + 128, 0, 1);
    acc0 = __builtin_amdgcn_wmma_f32_16x16x32_bf16(false, a0, false, bv,
                                                   (short)0, acc0, false, false);
    acc1 = __builtin_amdgcn_wmma_f32_16x16x32_bf16(false, a1, false, bv,
                                                   (short)0, acc1, false, false);
  }

  int feat = nt * 16 + lr;
#pragma unroll
  for (int half = 0; half < 2; ++half) {
    v8f acc = half ? acc1 : acc0;
    int nodeBase = mt * 32 + half * 16 + hb * 8;
#pragma unroll
    for (int i = 0; i < 8; ++i) {
      int node = nodeBase + i;
      size_t nmIdx = ((size_t)b * NB + node) * f_pad + feat;
      float v = alpha * acc[i];
      if (HAS_BETA) v += beta * betaSrc[nmIdx];
      out_nm[nmIdx] = v;
      __bf16 vb = (__bf16)v;
      if (feat < f)
        xcat[((size_t)b * NB + node) * Kp + feat * MDIFF + m] = vb;
      if (HAS_XTOUT)  // pad feats store 0 (operand pads are 0, beta unused)
        XtOut[((size_t)b * f_pad + feat) * NB + node] = vb;
    }
  }
}

// ---------------------------------------------------------------------------
// Weight GEMM: out(16384 x outF) = act( xcat(16384 x Kp) @ W + bias )
//   xcat bf16 node-major (A-op),  Wt bf16 [outF][Kp] (B-op, pre-transposed)
// One wave = 32x16 D tile.  act: 0 = sigmoid (gate), 1 = tanh (candidate)
// ---------------------------------------------------------------------------
__global__ __launch_bounds__(256) void gemm_w_kernel(
    const __bf16* __restrict__ Xc, const __bf16* __restrict__ Wt,
    const float* __restrict__ bias, float* __restrict__ out, int Kp, int outF,
    int act, int totalTiles) {
  int wid = (blockIdx.x * blockDim.x + threadIdx.x) >> 5;
  if (wid >= totalTiles) return;
  int lane = threadIdx.x & 31;
  int hb = lane >> 4, lr = lane & 15;
  int ntf = outF >> 4;
  int mt = wid / ntf, nt = wid % ntf;  // mt: 512 row-tiles of 32

  const __bf16* pa0 = Xc + ((size_t)(mt * 32 + lr)) * Kp + hb * 8;
  const __bf16* pa1 = pa0 + (size_t)16 * Kp;
  const __bf16* pb = Wt + ((size_t)(nt * 16 + lr)) * Kp + hb * 8;

  v8f acc0 = {0.f, 0.f, 0.f, 0.f, 0.f, 0.f, 0.f, 0.f};
  v8f acc1 = {0.f, 0.f, 0.f, 0.f, 0.f, 0.f, 0.f, 0.f};
#pragma unroll 2
  for (int k = 0; k < Kp; k += 32) {
    v16bf bv = cat8(*(const v8bf*)(pb + k), *(const v8bf*)(pb + k + 16));
    v16bf a0 = cat8(*(const v8bf*)(pa0 + k), *(const v8bf*)(pa0 + k + 16));
    v16bf a1 = cat8(*(const v8bf*)(pa1 + k), *(const v8bf*)(pa1 + k + 16));
    acc0 = __builtin_amdgcn_wmma_f32_16x16x32_bf16(false, a0, false, bv,
                                                   (short)0, acc0, false, false);
    acc1 = __builtin_amdgcn_wmma_f32_16x16x32_bf16(false, a1, false, bv,
                                                   (short)0, acc1, false, false);
  }
  int col = nt * 16 + lr;
  float bsv = bias[col];
#pragma unroll
  for (int half = 0; half < 2; ++half) {
    v8f acc = half ? acc1 : acc0;
    int rowBase = mt * 32 + half * 16 + hb * 8;
#pragma unroll
    for (int i = 0; i < 8; ++i) {
      float v = acc[i] + bsv;
      v = (act == 0) ? (1.f / (1.f + expf(-v))) : tanhf(v);
      out[(size_t)(rowBase + i) * outF + col] = v;
    }
  }
}

// ---------------------------------------------------------------------------
// GRU elementwise pieces
// ---------------------------------------------------------------------------
__global__ __launch_bounds__(256) void gate_split_kernel(
    const float* __restrict__ gate, const float* __restrict__ h,
    float* __restrict__ st2) {
  int idx = blockIdx.x * blockDim.x + threadIdx.x;
  if (idx >= BN * UNITS) return;
  int bn = idx / UNITS, u = idx % UNITS;
  st2[idx] = gate[(size_t)bn * 128 + u] * h[idx];  // r * hx
}

__global__ __launch_bounds__(256) void update_h_kernel(
    const float* __restrict__ gate, const float* __restrict__ cand,
    float* __restrict__ h) {
  int idx = blockIdx.x * blockDim.x + threadIdx.x;
  if (idx >= BN * UNITS) return;
  int bn = idx / UNITS, u = idx % UNITS;
  float uu = gate[(size_t)bn * 128 + 64 + u];
  h[idx] = uu * h[idx] + (1.f - uu) * cand[idx];
}

__global__ __launch_bounds__(256) void add_vec_kernel(float* __restrict__ a,
                                                      const float* __restrict__ b,
                                                      long n) {
  long idx = (long)blockIdx.x * blockDim.x + threadIdx.x;
  if (idx < n) a[idx] += b[idx];
}

// fc_his MLP (one-time, tiny vs GEMMs): relu(relu(H@fc1+b1)@fc2+b2)
__global__ __launch_bounds__(256) void fc1_kernel(
    const float* __restrict__ hid, const float* __restrict__ W,
    const float* __restrict__ bias, float* __restrict__ out) {
  int idx = blockIdx.x * blockDim.x + threadIdx.x;
  if (idx >= BN * 256) return;
  int bn = idx / 256, j = idx % 256;
  float acc = bias[j];
  for (int k = 0; k < 96; ++k) acc += hid[(size_t)bn * 96 + k] * W[k * 256 + j];
  out[idx] = fmaxf(acc, 0.f);
}

__global__ __launch_bounds__(256) void fc2_kernel(
    const float* __restrict__ t256, const float* __restrict__ W,
    const float* __restrict__ bias, float* __restrict__ out) {
  int idx = blockIdx.x * blockDim.x + threadIdx.x;
  if (idx >= BN * 64) return;
  int bn = idx / 64, j = idx % 64;
  float acc = bias[j];
  for (int k = 0; k < 256; ++k)
    acc += t256[(size_t)bn * 256 + k] * W[k * 64 + j];
  out[idx] = fmaxf(acc, 0.f);
}

// Projection (UNITS -> 1) + write output slice (B, N, HORIZON)
__global__ __launch_bounds__(256) void proj_kernel(
    const float* __restrict__ hL, const float* __restrict__ pW,
    const float* __restrict__ pb, float* __restrict__ dec_in,
    float* __restrict__ dout, int t) {
  int bn = blockIdx.x * blockDim.x + threadIdx.x;
  if (bn >= BN) return;
  float acc = pb[0];
  for (int k = 0; k < UNITS; ++k) acc += hL[(size_t)bn * UNITS + k] * pW[k];
  dec_in[bn] = acc;
  dout[(size_t)bn * HORIZON + t] = acc;
}

// ---------------------------------------------------------------------------
// Host orchestration
// ---------------------------------------------------------------------------
struct LP {
  int cin, f, f_pad, fM, Kp;
};

extern "C" void kernel_launch(void* const* d_in, const int* in_sizes, int n_in,
                              void* d_out, int out_size, void* d_ws,
                              size_t ws_size, hipStream_t stream) {
  (void)in_sizes; (void)n_in; (void)out_size; (void)ws_size;
  const float* hist = (const float*)d_in[0];
  const float* hidden = (const float*)d_in[1];
  const float* adj = (const float*)d_in[2];
  const float* Wp[4][4];  // [enc0,enc1,dec0,dec1][Wg,bg,Wc,bc]
  for (int l = 0; l < 4; ++l)
    for (int j = 0; j < 4; ++j) Wp[l][j] = (const float*)d_in[3 + l * 4 + j];
  const float* projW = (const float*)d_in[19];
  const float* projB = (const float*)d_in[20];
  const float* fc1W = (const float*)d_in[21];
  const float* fc1B = (const float*)d_in[22];
  const float* fc2W = (const float*)d_in[23];
  const float* fc2B = (const float*)d_in[24];
  float* dout = (float*)d_out;

  // ---- workspace bump allocator (~160 MB) ----
  char* ws = (char*)d_ws;
  size_t off = 0;
  auto alloc = [&](size_t bytes) -> char* {
    char* p = ws + off;
    off += (bytes + 255) & ~(size_t)255;
    return p;
  };
  const size_t SZA = (size_t)BB * NB * NB;
  __bf16* A1 = (__bf16*)alloc(SZA * 2);
  __bf16* A2 = (__bf16*)alloc(SZA * 2);
  float* dinv1 = (float*)alloc(BN * 4);
  float* dinv2 = (float*)alloc(BN * 4);
  float* h0 = (float*)alloc((size_t)BN * UNITS * 4);
  float* h1 = (float*)alloc((size_t)BN * UNITS * 4);
  float* his = (float*)alloc((size_t)BN * UNITS * 4);
  float* tmp256 = (float*)alloc((size_t)BN * 256 * 4);
  float* xnm0 = (float*)alloc((size_t)BN * 128 * 4);
  float* ynm1 = (float*)alloc((size_t)BN * 128 * 4);
  float* ynm2 = (float*)alloc((size_t)BN * 128 * 4);
  float* ynm3 = (float*)alloc((size_t)BN * 128 * 4);
  float* ynm4 = (float*)alloc((size_t)BN * 128 * 4);
  __bf16* Xt0 = (__bf16*)alloc((size_t)BB * 128 * NB * 2);
  __bf16* Xt1 = (__bf16*)alloc((size_t)BB * 128 * NB * 2);
  __bf16* Xt3 = (__bf16*)alloc((size_t)BB * 128 * NB * 2);
  __bf16* xcat = (__bf16*)alloc((size_t)BN * 640 * 2);
  __bf16* WtG[4];
  __bf16* WtC[4];
  for (int l = 0; l < 4; ++l) {
    WtG[l] = (__bf16*)alloc((size_t)128 * 640 * 2);
    WtC[l] = (__bf16*)alloc((size_t)64 * 640 * 2);
  }
  float* gate = (float*)alloc((size_t)BN * 128 * 4);
  float* cand = (float*)alloc((size_t)BN * UNITS * 4);
  float* st2 = (float*)alloc((size_t)BN * UNITS * 4);
  float* decin = (float*)alloc((size_t)BN * 4);

  const LP lps[4] = {{2, 66, 80, 330, 352},
                     {64, 128, 128, 640, 640},
                     {1, 65, 80, 325, 352},
                     {64, 128, 128, 640, 640}};

  auto nblk = [](long n) { return (unsigned)((n + 255) / 256); };

  // ---- per-launch setup (deterministic, no caching) ----
  dinv_kernel<<<nblk(BN), 256, 0, stream>>>(adj, dinv1, dinv2);
  build_supports_kernel<<<nblk((long)SZA), 256, 0, stream>>>(adj, dinv1, dinv2,
                                                             A1, A2);
  for (int l = 0; l < 4; ++l) {
    build_wt_kernel<<<nblk(128 * lps[l].Kp), 256, 0, stream>>>(
        Wp[l][0], WtG[l], lps[l].fM, 128, lps[l].Kp);
    build_wt_kernel<<<nblk(64 * lps[l].Kp), 256, 0, stream>>>(
        Wp[l][2], WtC[l], lps[l].fM, 64, lps[l].Kp);
  }
  zero_f32_kernel<<<nblk((long)BN * UNITS), 256, 0, stream>>>(h0,
                                                              (long)BN * UNITS);
  zero_f32_kernel<<<nblk((long)BN * UNITS), 256, 0, stream>>>(h1,
                                                              (long)BN * UNITS);
  zero_f32_kernel<<<nblk(BN), 256, 0, stream>>>(decin, BN);

  // fc_his (independent of the encoder scan)
  fc1_kernel<<<nblk((long)BN * 256), 256, 0, stream>>>(hidden, fc1W, fc1B,
                                                       tmp256);
  fc2_kernel<<<nblk((long)BN * 64), 256, 0, stream>>>(tmp256, fc2W, fc2B, his);

  // ---- gconv: 4 chained diffusion GEMMs + 1 weight GEMM ----
  auto run_gconv = [&](const float* xin, int bstride, const float* state,
                       const __bf16* Wt, const float* bias, int outF, int act,
                       float* outBuf, const LP& lp) {
    prepare_x0_kernel<<<nblk((long)BN * lp.f_pad), 256, 0, stream>>>(
        xin, bstride, lp.cin, state, xnm0, Xt0, xcat, lp.Kp, lp.f, lp.f_pad);
    int padc = lp.Kp - lp.fM;
    if (padc > 0)
      zero_xcat_pad_kernel<<<nblk((long)BN * padc), 256, 0, stream>>>(
          xcat, lp.Kp, lp.fM, padc);
    int tiles = BB * 16 * (lp.f_pad / 16);  // 32-row x 16-col tiles
    unsigned gb = nblk((long)tiles * 32);
    // m=1: y1 = A1 x0
    gemm_diff_kernel<false, true><<<gb, 256, 0, stream>>>(
        A1, Xt0, ynm1, nullptr, 1.f, 0.f, xcat, lp.Kp, 1, lp.f, lp.f_pad, Xt1,
        tiles);
    // m=2: y2 = 2 A1 y1 - x0
    gemm_diff_kernel<true, false><<<gb, 256, 0, stream>>>(
        A1, Xt1, ynm2, xnm0, 2.f, -1.f, xcat, lp.Kp, 2, lp.f, lp.f_pad, nullptr,
        tiles);
    // m=3: y3 = A2 y1   (faithful to reference's x0-carry across supports)
    gemm_diff_kernel<false, true><<<gb, 256, 0, stream>>>(
        A2, Xt1, ynm3, nullptr, 1.f, 0.f, xcat, lp.Kp, 3, lp.f, lp.f_pad, Xt3,
        tiles);
    // m=4: y4 = 2 A2 y3 - y1
    gemm_diff_kernel<true, false><<<gb, 256, 0, stream>>>(
        A2, Xt3, ynm4, ynm1, 2.f, -1.f, xcat, lp.Kp, 4, lp.f, lp.f_pad, nullptr,
        tiles);
    int tilesW = (BN / 32) * (outF / 16);
    gemm_w_kernel<<<nblk((long)tilesW * 32), 256, 0, stream>>>(
        xcat, Wt, bias, outBuf, lp.Kp, outF, act, tilesW);
  };

  auto run_cell = [&](const float* xin, int bstride, float* h, int li) {
    const LP& lp = lps[li];
    run_gconv(xin, bstride, h, WtG[li], Wp[li][1], 128, 0, gate, lp);
    gate_split_kernel<<<nblk((long)BN * UNITS), 256, 0, stream>>>(gate, h, st2);
    run_gconv(xin, bstride, st2, WtC[li], Wp[li][3], 64, 1, cand, lp);
    update_h_kernel<<<nblk((long)BN * UNITS), 256, 0, stream>>>(gate, cand, h);
  };

  // ---- encoder scan ----
  for (int t = 0; t < 12; ++t) {
    run_cell(hist + (size_t)t * NB * 2, 12 * NB * 2, h0, 0);  // layer 0
    run_cell(h0, NB * UNITS, h1, 1);                          // layer 1
  }
  add_vec_kernel<<<nblk((long)BN * UNITS), 256, 0, stream>>>(h0, his,
                                                             (long)BN * UNITS);
  add_vec_kernel<<<nblk((long)BN * UNITS), 256, 0, stream>>>(h1, his,
                                                             (long)BN * UNITS);

  // ---- decoder scan ----
  for (int t = 0; t < HORIZON; ++t) {
    run_cell(decin, NB, h0, 2);
    run_cell(h0, NB * UNITS, h1, 3);
    proj_kernel<<<nblk(BN), 256, 0, stream>>>(h1, projW, projB, decin, dout, t);
  }
}
